// LocalitySelfAttention_18356690224042
// MI455X (gfx1250) — compile-verified, matrix operand order checked by reading, not run on hardware
//
#include <hip/hip_runtime.h>
#include <hip/hip_bf16.h>

typedef __attribute__((ext_vector_type(16))) __bf16 v16bf;
typedef __attribute__((ext_vector_type(8)))  float  v8f;

constexpr int Bv = 4, Nv = 2048, Cv = 768, Hv = 12, Dv = 64;
constexpr int M_TOT = Bv * Nv;       // 8192
constexpr int QKV_N = 3 * Cv;        // 2304
constexpr float SCALE = 0.125f;      // 64^-0.5

union FragBF {
  v16bf v;
  uint4 q[2];
};

__device__ __forceinline__ v8f wmma_bf16(const FragBF& a, const FragBF& b, v8f c) {
  return __builtin_amdgcn_wmma_f32_16x16x32_bf16(false, a.v, false, b.v, (short)0,
                                                 c, false, false);
}

// CDNA5 async global->LDS copy (ASYNCcnt-tracked, no VGPR data path).
// LDS offset = low 32 bits of the generic pointer (aperture-based flat addr).
__device__ __forceinline__ uint32_t lds_off(const void* p) {
  return (uint32_t)(uintptr_t)p;
}
__device__ __forceinline__ void async_copy_b128(uint32_t ldsoff, const void* gptr) {
  asm volatile("global_load_async_to_lds_b128 %0, %1, off"
               :: "v"(ldsoff), "v"((unsigned long long)(uintptr_t)gptr)
               : "memory");
}
__device__ __forceinline__ void wait_async0() {
  asm volatile("s_wait_asynccnt 0x0" ::: "memory");
}
__device__ __forceinline__ void wait_async1() {   // allow 1 copy in flight
  asm volatile("s_wait_asynccnt 0x1" ::: "memory");
}

// ---------------------------------------------------------------------------
// Kernel 1: QKV = X(8192x768) @ Wqkv(768x2304), fp32 in -> bf16 out.
// Q,K stored [B,H,N,D]; V stored transposed [B,H,D,N] (feeds async V staging).
// Q pre-scaled by SCALE. WG tile 128x128, 8 waves (2x4), K-step 32.
// ---------------------------------------------------------------------------
__global__ __launch_bounds__(256) void qkv_gemm_kernel(
    const float* __restrict__ x, const float* __restrict__ wqkv,
    __bf16* __restrict__ Qb, __bf16* __restrict__ Kb, __bf16* __restrict__ Vb) {
  __shared__ __bf16 As[128][32];   // As[m][k]
  __shared__ __bf16 Bs[128][32];   // Bs[n][k] = W[k0+k][j0+n]

  const int tid = threadIdx.x;
  const int lane = tid & 31, wid = tid >> 5;
  const int g = lane >> 4, laneN = lane & 15;
  const int j0 = blockIdx.x * 128;
  const int m0 = blockIdx.y * 128;
  const int rm = (wid >> 2) * 64;  // wave row base (0 or 64)
  const int cn = (wid & 3) * 32;   // wave col base (0..96)

  v8f acc[4][2] = {};

  for (int k0 = 0; k0 < Cv; k0 += 32) {
    // stage A tile (convert fp32 -> bf16)
#pragma unroll
    for (int it = 0; it < 4; ++it) {
      int e = (tid + it * 256) * 4;
      int row = e >> 5, kk = e & 31;
      float4 f = *reinterpret_cast<const float4*>(&x[(size_t)(m0 + row) * Cv + k0 + kk]);
      union { __bf16 b[4]; uint2 u; } t;
      t.b[0] = (__bf16)f.x; t.b[1] = (__bf16)f.y;
      t.b[2] = (__bf16)f.z; t.b[3] = (__bf16)f.w;
      *reinterpret_cast<uint2*>(&As[row][kk]) = t.u;
    }
    // stage B tile transposed
#pragma unroll
    for (int it = 0; it < 4; ++it) {
      int e = (tid + it * 256) * 4;
      int k = e >> 7, n = e & 127;
      float4 f = *reinterpret_cast<const float4*>(&wqkv[(size_t)(k0 + k) * QKV_N + j0 + n]);
      Bs[n + 0][k] = (__bf16)f.x;
      Bs[n + 1][k] = (__bf16)f.y;
      Bs[n + 2][k] = (__bf16)f.z;
      Bs[n + 3][k] = (__bf16)f.w;
    }
    __syncthreads();

    FragBF bfr[2];
#pragma unroll
    for (int nt = 0; nt < 2; ++nt) {
      const __bf16* p = &Bs[cn + nt * 16 + laneN][g * 8];
      bfr[nt].q[0] = *reinterpret_cast<const uint4*>(p);
      bfr[nt].q[1] = *reinterpret_cast<const uint4*>(p + 16);
    }
#pragma unroll
    for (int mt = 0; mt < 4; ++mt) {
      FragBF afr;
      const __bf16* p = &As[rm + mt * 16 + laneN][g * 8];
      afr.q[0] = *reinterpret_cast<const uint4*>(p);
      afr.q[1] = *reinterpret_cast<const uint4*>(p + 16);
#pragma unroll
      for (int nt = 0; nt < 2; ++nt)
        acc[mt][nt] = wmma_bf16(afr, bfr[nt], acc[mt][nt]);
    }
    __syncthreads();
  }

  // epilogue: scatter into Q/K [B,H,N,D] (b16) and V [B,H,D,N] (packed b128)
  const int bb = m0 >> 11;   // batch is uniform within a 128-row block
#pragma unroll
  for (int nt = 0; nt < 2; ++nt) {
    const int j = j0 + cn + nt * 16 + laneN;
    const int s = j / Cv;              // uniform per wave per nt
    const int rem = j % Cv;
    const int hh = rem >> 6, dd = rem & 63;
    if (s == 2) {
      // V transposed: 8 consecutive n per lane -> one 16B store
      __bf16* vbase = Vb + (((size_t)(bb * Hv + hh)) * Dv + dd) * Nv;
#pragma unroll
      for (int mt = 0; mt < 4; ++mt) {
        const int nn = (m0 + rm + mt * 16 + 8 * g) & (Nv - 1);
        union { __bf16 b[8]; uint4 u; } t;
#pragma unroll
        for (int v = 0; v < 8; ++v) t.b[v] = (__bf16)acc[mt][nt][v];
        *reinterpret_cast<uint4*>(&vbase[nn]) = t.u;
      }
    } else {
      __bf16* dst = (s == 0) ? Qb : Kb;
      const float sc = (s == 0) ? SCALE : 1.0f;
#pragma unroll
      for (int mt = 0; mt < 4; ++mt) {
#pragma unroll
        for (int v = 0; v < 8; ++v) {
          const int m = m0 + rm + mt * 16 + 8 * g + v;
          const int nn = m & (Nv - 1);
          dst[(((size_t)(bb * Hv + hh)) * Nv + nn) * Dv + dd] =
              (__bf16)(acc[mt][nt][v] * sc);
        }
      }
    }
  }
}

// ---------------------------------------------------------------------------
// Kernel 2: flash attention per (b,h). WG = 128 Q rows (8 waves x 16 rows).
// KV blocks of 32. V^T double-buffered via global_load_async_to_lds_b128 so
// the DMA of block i+1 overlaps the WMMA/softmax work of block i.
// ---------------------------------------------------------------------------
__global__ __launch_bounds__(256) void attn_kernel(
    const __bf16* __restrict__ Qb, const __bf16* __restrict__ Kb,
    const __bf16* __restrict__ Vb, const float* __restrict__ temperature,
    __bf16* __restrict__ Ob) {
  __shared__ __bf16 Vt[2][64][32];    // double-buffered V^T tile
  __shared__ __bf16 Ps[8][16][32];    // per-wave P staging

  const int tid = threadIdx.x, lane = tid & 31, wid = tid >> 5;
  const int g = lane >> 4, laneN = lane & 15;
  const int bh = blockIdx.y;               // 0..47
  const int b = bh / Hv, h = bh % Hv;
  const float omt = 1.0f - temperature[h];
  const int r0 = blockIdx.x * 128 + wid * 16;

  const __bf16* qb  = Qb + (size_t)bh * Nv * Dv;
  const __bf16* kb  = Kb + (size_t)bh * Nv * Dv;
  const __bf16* vtg = Vb + (size_t)bh * Dv * Nv;   // [D][N]

  // resident Q fragments (16 rows x 64 cols = 2 A-frags)
  FragBF qf[2];
  {
    const __bf16* p = qb + (size_t)(r0 + laneN) * Dv;
    qf[0].q[0] = *reinterpret_cast<const uint4*>(p + g * 8);
    qf[0].q[1] = *reinterpret_cast<const uint4*>(p + g * 8 + 16);
    qf[1].q[0] = *reinterpret_cast<const uint4*>(p + 32 + g * 8);
    qf[1].q[1] = *reinterpret_cast<const uint4*>(p + 32 + g * 8 + 16);
  }

  const int vd = tid >> 2, vk = (tid & 3) * 8;   // async V staging coords
  const __bf16* vrow = vtg + (size_t)vd * Nv + vk;

  float mrow[8], lrow[8];
#pragma unroll
  for (int v = 0; v < 8; ++v) { mrow[v] = -3.0e38f; lrow[v] = 0.0f; }
  v8f oacc[4] = {};

  // prologue: start DMA of KV block 0
  async_copy_b128(lds_off(&Vt[0][vd][vk]), vrow);

  for (int j0 = 0; j0 < Nv; j0 += 32) {
    const int cur = (j0 >> 5) & 1;
    if (j0 + 32 < Nv) {
      // kick DMA for the next block into the other buffer, keep it in flight
      async_copy_b128(lds_off(&Vt[cur ^ 1][vd][vk]), vrow + j0 + 32);
      wait_async1();      // oldest copy (current buffer) is complete
    } else {
      wait_async0();
    }
    __syncthreads();      // current V^T buffer visible to all waves

    if (j0 + 32 < Nv)
      __builtin_prefetch(kb + (size_t)(j0 + 32 + laneN) * Dv, 0, 0);

    // S = (Q*scale) K^T over 32 columns: two 16x16 tiles
    v8f s0 = {}, s1 = {};
    {
      FragBF kf;
      const __bf16* p = kb + (size_t)(j0 + laneN) * Dv;
      kf.q[0] = *reinterpret_cast<const uint4*>(p + g * 8);
      kf.q[1] = *reinterpret_cast<const uint4*>(p + g * 8 + 16);
      s0 = wmma_bf16(qf[0], kf, s0);
      kf.q[0] = *reinterpret_cast<const uint4*>(p + 32 + g * 8);
      kf.q[1] = *reinterpret_cast<const uint4*>(p + 32 + g * 8 + 16);
      s0 = wmma_bf16(qf[1], kf, s0);
      const __bf16* p2 = kb + (size_t)(j0 + 16 + laneN) * Dv;
      kf.q[0] = *reinterpret_cast<const uint4*>(p2 + g * 8);
      kf.q[1] = *reinterpret_cast<const uint4*>(p2 + g * 8 + 16);
      s1 = wmma_bf16(qf[0], kf, s1);
      kf.q[0] = *reinterpret_cast<const uint4*>(p2 + 32 + g * 8);
      kf.q[1] = *reinterpret_cast<const uint4*>(p2 + 32 + g * 8 + 16);
      s1 = wmma_bf16(qf[1], kf, s1);
    }

    // diagonal temperature + row max
    const int col0 = j0 + laneN, col1 = col0 + 16;
    float pmax[8];
#pragma unroll
    for (int v = 0; v < 8; ++v) {
      const int r = r0 + 8 * g + v;
      if (r == col0) s0[v] *= omt;
      if (r == col1) s1[v] *= omt;
      pmax[v] = fmaxf(s0[v], s1[v]);
    }
#pragma unroll
    for (int v = 0; v < 8; ++v) {
#pragma unroll
      for (int off = 1; off < 16; off <<= 1)
        pmax[v] = fmaxf(pmax[v], __shfl_xor(pmax[v], off, 32));
    }

    // online softmax update
    float psum[8];
#pragma unroll
    for (int v = 0; v < 8; ++v) {
      const float mnew = fmaxf(mrow[v], pmax[v]);
      const float corr = __expf(mrow[v] - mnew);
      const float p0 = __expf(s0[v] - mnew);
      const float p1 = __expf(s1[v] - mnew);
      s0[v] = p0; s1[v] = p1;
      psum[v] = p0 + p1;
      mrow[v] = mnew;
      lrow[v] *= corr;
#pragma unroll
      for (int t = 0; t < 4; ++t) oacc[t][v] *= corr;
    }
#pragma unroll
    for (int v = 0; v < 8; ++v) {
#pragma unroll
      for (int off = 1; off < 16; off <<= 1)
        psum[v] += __shfl_xor(psum[v], off, 32);
      lrow[v] += psum[v];
    }

    // P (C-layout) -> LDS -> A-fragment layout
#pragma unroll
    for (int v = 0; v < 8; ++v) {
      const int r16 = 8 * g + v;
      Ps[wid][r16][laneN] = (__bf16)s0[v];
      Ps[wid][r16][laneN + 16] = (__bf16)s1[v];
    }
    asm volatile("s_wait_dscnt 0" ::: "memory");

    FragBF pf;
    {
      const __bf16* p = &Ps[wid][laneN][g * 8];
      pf.q[0] = *reinterpret_cast<const uint4*>(p);
      pf.q[1] = *reinterpret_cast<const uint4*>(p + 16);
    }
#pragma unroll
    for (int t = 0; t < 4; ++t) {
      FragBF vf;
      const __bf16* p = &Vt[cur][t * 16 + laneN][g * 8];
      vf.q[0] = *reinterpret_cast<const uint4*>(p);
      vf.q[1] = *reinterpret_cast<const uint4*>(p + 16);
      oacc[t] = wmma_bf16(pf, vf, oacc[t]);
    }
    __syncthreads();      // all reads of Vt[cur] done before it is refilled
  }

  // normalize + store O as bf16 [B,N,C]
  float inv[8];
#pragma unroll
  for (int v = 0; v < 8; ++v) inv[v] = 1.0f / lrow[v];
#pragma unroll
  for (int t = 0; t < 4; ++t) {
    const int c = h * 64 + t * 16 + laneN;
#pragma unroll
    for (int v = 0; v < 8; ++v) {
      const int r = r0 + 8 * g + v;
      Ob[((size_t)(b * Nv + r)) * Cv + c] = (__bf16)(oacc[t][v] * inv[v]);
    }
  }
}

// ---------------------------------------------------------------------------
// Kernel 3: OUT = O(8192x768 bf16) @ Wproj(768x768 fp32->bf16) + bias, fp32 out
// A-tile staged via async global->LDS b128 copies overlapped with B convert.
// ---------------------------------------------------------------------------
__global__ __launch_bounds__(256) void proj_gemm_kernel(
    const __bf16* __restrict__ Ob, const float* __restrict__ wproj,
    const float* __restrict__ bproj, float* __restrict__ out) {
  __shared__ __bf16 As[128][32];
  __shared__ __bf16 Bs[128][32];

  const int tid = threadIdx.x;
  const int lane = tid & 31, wid = tid >> 5;
  const int g = lane >> 4, laneN = lane & 15;
  const int j0 = blockIdx.x * 128;
  const int m0 = blockIdx.y * 128;
  const int rm = (wid >> 2) * 64;
  const int cn = (wid & 3) * 32;

  v8f acc[4][2] = {};

  for (int k0 = 0; k0 < Cv; k0 += 32) {
    // async A staging: 512 x b128 chunks, 2 per thread (in flight during B cvt)
#pragma unroll
    for (int it = 0; it < 2; ++it) {
      int c = tid + it * 256;
      int row = c >> 2, kk = (c & 3) * 8;
      async_copy_b128(lds_off(&As[row][kk]),
                      &Ob[(size_t)(m0 + row) * Cv + k0 + kk]);
    }
    // B staging with fp32 -> bf16 conversion (VALU path, overlaps the DMA)
#pragma unroll
    for (int it = 0; it < 4; ++it) {
      int e = (tid + it * 256) * 4;
      int k = e >> 7, n = e & 127;
      float4 f = *reinterpret_cast<const float4*>(&wproj[(size_t)(k0 + k) * Cv + j0 + n]);
      Bs[n + 0][k] = (__bf16)f.x;
      Bs[n + 1][k] = (__bf16)f.y;
      Bs[n + 2][k] = (__bf16)f.z;
      Bs[n + 3][k] = (__bf16)f.w;
    }
    wait_async0();
    __syncthreads();

    FragBF bfr[2];
#pragma unroll
    for (int nt = 0; nt < 2; ++nt) {
      const __bf16* p = &Bs[cn + nt * 16 + laneN][g * 8];
      bfr[nt].q[0] = *reinterpret_cast<const uint4*>(p);
      bfr[nt].q[1] = *reinterpret_cast<const uint4*>(p + 16);
    }
#pragma unroll
    for (int mt = 0; mt < 4; ++mt) {
      FragBF afr;
      const __bf16* p = &As[rm + mt * 16 + laneN][g * 8];
      afr.q[0] = *reinterpret_cast<const uint4*>(p);
      afr.q[1] = *reinterpret_cast<const uint4*>(p + 16);
#pragma unroll
      for (int nt = 0; nt < 2; ++nt)
        acc[mt][nt] = wmma_bf16(afr, bfr[nt], acc[mt][nt]);
    }
    __syncthreads();
  }

#pragma unroll
  for (int nt = 0; nt < 2; ++nt) {
    const int j = j0 + cn + nt * 16 + laneN;
    const float bias = bproj[j];
#pragma unroll
    for (int mt = 0; mt < 4; ++mt) {
#pragma unroll
      for (int v = 0; v < 8; ++v) {
        const int m = m0 + rm + mt * 16 + 8 * g + v;
        out[(size_t)m * Cv + j] = acc[mt][nt][v] + bias;
      }
    }
  }
}

// ---------------------------------------------------------------------------
extern "C" void kernel_launch(void* const* d_in, const int* in_sizes, int n_in,
                              void* d_out, int out_size, void* d_ws, size_t ws_size,
                              hipStream_t stream) {
  (void)in_sizes; (void)n_in; (void)out_size; (void)ws_size;
  const float* x     = (const float*)d_in[0];
  const float* wqkv  = (const float*)d_in[1];
  const float* wproj = (const float*)d_in[2];
  const float* bproj = (const float*)d_in[3];
  const float* temp  = (const float*)d_in[4];

  char* ws = (char*)d_ws;
  const size_t sz = (size_t)M_TOT * Cv * sizeof(__bf16);  // 12.58 MB each
  __bf16* Qb = (__bf16*)(ws);                 // [B,H,N,D]
  __bf16* Kb = (__bf16*)(ws + sz);            // [B,H,N,D]
  __bf16* Vb = (__bf16*)(ws + 2 * sz);        // [B,H,D,N] (transposed)
  __bf16* Ob = (__bf16*)(ws + 3 * sz);        // [B,N,C]

  qkv_gemm_kernel<<<dim3(QKV_N / 128, M_TOT / 128), 256, 0, stream>>>(x, wqkv, Qb, Kb, Vb);
  attn_kernel<<<dim3(Nv / 128, Bv * Hv), 256, 0, stream>>>(Qb, Kb, Vb, temp, Ob);
  proj_gemm_kernel<<<dim3(Cv / 128, M_TOT / 128), 256, 0, stream>>>(Ob, wproj, bproj, (float*)d_out);
}
